// MatrixRouting_66864050864710
// MI455X (gfx1250) — matrix operand order repaired and checked by reading, not compiled, not run on hardware
//
#include <hip/hip_runtime.h>

typedef _Float16 v16h __attribute__((ext_vector_type(16)));
typedef float    v8f  __attribute__((ext_vector_type(8)));

#define BB 4
#define II 1152
#define CC 1152
#define OO 32
#define WW2 36
#define PP 16
#define DD 17
#define NROUT 3
#define EPSF 1e-10f
#define LAMBDAF 1e-4f
#define HALF_LN_2PI 0.91893853320467274178f

// -------------------------------------------------------------------------
// M-step: one wave per (b,c). Accumulate S0, S1_p = sum_i r*v_p,
// S2_p = sum_i r*v_p^2 using v_wmma_f32_16x16x32_f16.
//   B operand: lane L holds V column p = L%16, 16 f16 K rows (table order).
//   A operand: only row 0 nonzero = r_k; per 16-bit A layout, row 0 lives in
//   lane 0 (K=0..7,16..23) and lane 16 (K=8..15,24..31).
//   D row 0 (lanes 0-15, element [0]) = the 16 p-sums.
// -------------------------------------------------------------------------
__global__ __launch_bounds__(256) void mstep_kernel(
    const float* __restrict__ votes, const float* __restrict__ R,
    float* __restrict__ S0, float* __restrict__ S1, float* __restrict__ S2,
    int iter)
{
  const int lane = threadIdx.x & 31;
  const int wave = threadIdx.x >> 5;
  const int pair = blockIdx.x * 8 + wave;   // 0 .. B*C-1
  const int b = pair / CC;
  const int c = pair % CC;

  v8f acc1 = {};   // S1 accumulator (f32)
  v8f acc2 = {};   // S2 accumulator (f32)
  float s0 = 0.0f;

  for (int chunk = 0; chunk < II / 32; ++chunk) {
    const int i = chunk * 32 + lane;
    const long base = ((long)(b * II + i) * CC + c) * DD;

    float v[PP];
#pragma unroll
    for (int p = 0; p < PP; ++p) v[p] = votes[base + p];
    const float act = votes[base + PP];

    float rr = (iter == 0) ? (1.0f / OO) : R[(long)(b * II + i) * CC + c];
    rr *= act;
    s0 += rr;

    v16h bv, bv2;
#pragma unroll
    for (int p = 0; p < PP; ++p) {
      const int kk = (p < 8) ? (8 * (lane >> 4) + p) : (16 + 8 * (lane >> 4) + (p - 8));
      const float x = votes[((long)(b * II + chunk * 32 + kk) * CC + c) * DD + (lane & 15)];
      bv[p]  = (_Float16)x;
      bv2[p] = (_Float16)(x * x);
    }

    // Build A (row 0 = r_k, other rows zero) per 16-bit A-matrix layout.
    v16h av;
#pragma unroll
    for (int j = 0; j < 16; ++j) {
      int src = (lane < 16) ? ((j < 8) ? j : j + 8)
                            : ((j < 8) ? j + 8 : j + 16);
      float rv = __shfl(rr, src, 32);
      av[j] = (lane == 0 || lane == 16) ? (_Float16)rv : (_Float16)0.0f;
    }

    acc1 = __builtin_amdgcn_wmma_f32_16x16x32_f16(
        false, av, false, bv,  (short)0, acc1, false, false);
    acc2 = __builtin_amdgcn_wmma_f32_16x16x32_f16(
        false, av, false, bv2, (short)0, acc2, false, false);
  }

  // Reduce S0 across the wave.
#pragma unroll
  for (int off = 16; off >= 1; off >>= 1) s0 += __shfl_xor(s0, off, 32);

  // D layout: VGPR0, lanes 0-15 -> M=0, N=lane.
  if (lane < 16) {
    S1[(long)pair * PP + lane] = acc1[0];
    S2[(long)pair * PP + lane] = acc2[0];
  }
  if (lane == 0) S0[pair] = s0;
}

// -------------------------------------------------------------------------
// Per-(b,c) statistics: mu, 1/(2*sigma^2), log_sigma + 0.5*ln(2pi), activation.
// -------------------------------------------------------------------------
__global__ __launch_bounds__(256) void stats_kernel(
    const float* __restrict__ S0, const float* __restrict__ S1,
    const float* __restrict__ S2,
    const float* __restrict__ beta_v, const float* __restrict__ beta_a,
    float* __restrict__ mu, float* __restrict__ i2s,
    float* __restrict__ coef, float* __restrict__ aout)
{
  const int idx = blockIdx.x * 256 + threadIdx.x;
  if (idx >= BB * CC) return;
  const int c = idx % CC;
  const int o = c / WW2;

  const float s0 = S0[idx];
  const float inv_s0 = 1.0f / s0;
  float sumlog = 0.0f;
#pragma unroll
  for (int p = 0; p < PP; ++p) {
    const float m   = S1[(long)idx * PP + p] * inv_s0;
    float var = S2[(long)idx * PP + p] * inv_s0 - m * m;
    var = fmaxf(var, 0.0f);
    const float lg = logf(sqrtf(var) + EPSF);
    mu  [(long)idx * PP + p] = m;
    i2s [(long)idx * PP + p] = 1.0f / (2.0f * var + 1e-30f);
    coef[(long)idx * PP + p] = lg + HALF_LN_2PI;
    sumlog += lg;
  }
  const float costsum = (16.0f * beta_v[o] + sumlog) * s0;
  const float z = LAMBDAF * (beta_a[o] - costsum);
  aout[idx] = 1.0f / (1.0f + __expf(-z));
}

// -------------------------------------------------------------------------
// E-step: one block per (b,i). 16 lanes per c compute exp(ln_p) and reduce
// over p; block reduces ap over all c in LDS; second sweep normalizes R.
// -------------------------------------------------------------------------
__global__ __launch_bounds__(256) void estep_kernel(
    const float* __restrict__ votes,
    const float* __restrict__ mu, const float* __restrict__ i2s,
    const float* __restrict__ coef, const float* __restrict__ aout,
    float* __restrict__ R)
{
  __shared__ float gsum[16];
  __shared__ float stot;
  const int t   = threadIdx.x;
  const int bi  = blockIdx.x;        // 0 .. B*I-1
  const int b   = bi / II;
  const int grp = t >> 4;            // 16 c-groups of 16 p-lanes
  const int p   = t & 15;
  const long vrow = (long)bi * CC * DD;
  const long rrow = (long)bi * CC;

  float gacc = 0.0f;
  for (int ct = 0; ct < CC; ct += 16) {
    const int c = ct + grp;
    const long sidx = ((long)b * CC + c) * PP + p;
    const float v  = votes[vrow + (long)c * DD + p];
    const float d  = v - mu[sidx];
    const float lp = -(d * d) * i2s[sidx] - coef[sidx];
    float pe = __expf(lp);
#pragma unroll
    for (int m = 8; m >= 1; m >>= 1) pe += __shfl_xor(pe, m, 32);
    if (p == 0) {
      const float ap = aout[b * CC + c] * pe;
      R[rrow + c] = ap;     // un-normalized
      gacc += ap;
    }
  }
  if (p == 0) gsum[grp] = gacc;
  __syncthreads();
  if (t == 0) {
    float tot = 0.0f;
#pragma unroll
    for (int g = 0; g < 16; ++g) tot += gsum[g];
    stot = tot + EPSF;
  }
  __syncthreads();
  const float inv = 1.0f / stot;
  for (int c = t; c < CC; c += 256)
    R[rrow + c] = R[rrow + c] * inv + EPSF;
}

// -------------------------------------------------------------------------
// Assemble output (B, O, w, w, 17): mu in [0..15], activation in [16].
// -------------------------------------------------------------------------
__global__ __launch_bounds__(256) void finalize_kernel(
    const float* __restrict__ mu, const float* __restrict__ aout,
    float* __restrict__ out)
{
  const int idx = blockIdx.x * 256 + threadIdx.x;
  if (idx >= BB * OO * WW2 * DD) return;
  const int d  = idx % DD;
  const int r  = idx / DD;
  const int ww = r % WW2;
  const int r2 = r / WW2;
  const int o  = r2 % OO;
  const int b  = r2 / OO;
  const int c  = o * WW2 + ww;
  out[idx] = (d < PP) ? mu[((long)b * CC + c) * PP + d] : aout[b * CC + c];
}

extern "C" void kernel_launch(void* const* d_in, const int* in_sizes, int n_in,
                              void* d_out, int out_size, void* d_ws, size_t ws_size,
                              hipStream_t stream)
{
  const float* votes  = (const float*)d_in[0];
  const float* beta_v = (const float*)d_in[1];
  const float* beta_a = (const float*)d_in[2];
  // output_dim=32, num_routing=3 fixed per setup_inputs().

  float* ws  = (float*)d_ws;
  float* R    = ws;                                  // B*I*C
  float* S0   = R    + (size_t)BB * II * CC;         // B*C
  float* S1   = S0   + (size_t)BB * CC;              // B*C*P
  float* S2   = S1   + (size_t)BB * CC * PP;         // B*C*P
  float* mu   = S2   + (size_t)BB * CC * PP;         // B*C*P
  float* i2s  = mu   + (size_t)BB * CC * PP;         // B*C*P
  float* coef = i2s  + (size_t)BB * CC * PP;         // B*C*P
  float* aout = coef + (size_t)BB * CC * PP;         // B*C
  float* out  = (float*)d_out;

  for (int it = 0; it < NROUT; ++it) {
    mstep_kernel<<<BB * CC / 8, 256, 0, stream>>>(votes, R, S0, S1, S2, it);
    stats_kernel<<<(BB * CC + 255) / 256, 256, 0, stream>>>(
        S0, S1, S2, beta_v, beta_a, mu, i2s, coef, aout);
    if (it != NROUT - 1)
      estep_kernel<<<BB * II, 256, 0, stream>>>(votes, mu, i2s, coef, aout, R);
  }
  finalize_kernel<<<(BB * OO * WW2 * DD + 255) / 256, 256, 0, stream>>>(mu, aout, out);
}
